// CLIPDecoder_83786222011049
// MI455X (gfx1250) — compile-verified, matrix operand order checked by reading, not run on hardware
//
#include <hip/hip_runtime.h>

// Problem constants (hardcoded from reference): L=S=E=1024, N=16
#define LQ 1024
#define SK 1024
#define NBATCH 16
#define EDIM 1024

typedef __attribute__((ext_vector_type(16))) __bf16 v16bf;
typedef __attribute__((ext_vector_type(8)))  float  v8f;

// GEMM block tiling: 256 threads = 8 waves in a 2(M) x 4(N) grid over a
// 128x128 block tile; each wave owns a 64x32 patch = 4x2 WMMA 16x16 tiles
// -> 8 v_wmma per K-step per wave.
constexpr int BM  = 128;
constexpr int BN  = 128;
constexpr int BK  = 32;
constexpr int PAD = 8;        // bf16 elements of row padding in LDS
constexpr int LDT = BK + PAD; // 40 elements = 80 bytes (16B aligned rows)
constexpr int TILE_ELEMS = (BM + BN) * LDT;           // one double-buffer slot
constexpr int TILE_BYTES = TILE_ELEMS * 2;            // 20480 B

// Native bf16 convert (emits v_cvt_pk_bf16_f32 pairs).
__device__ __forceinline__ unsigned short f2bf(float x) {
    union { __bf16 h; unsigned short u; } c;
    c.h = (__bf16)x;
    return c.u;
}

// Async global -> LDS, 16 bytes per lane (CDNA5 data path, ASYNCcnt).
__device__ __forceinline__ void async_b128(unsigned ldsAddr, const void* gptr) {
    asm volatile("global_load_async_to_lds_b128 %0, %1, off"
                 :: "v"(ldsAddr), "v"(gptr)
                 : "memory");
}
__device__ __forceinline__ void wait_async0() {
    asm volatile("s_wait_asynccnt 0x0" ::: "memory");
}
// <=4 outstanding: the 4 newest belong to the next tile, so the current
// tile's 4 asyncs (in-order ASYNCcnt) are complete.
__device__ __forceinline__ void wait_async4() {
    asm volatile("s_wait_asynccnt 0x4" ::: "memory");
}

// Load one 16x32 bf16 fragment from an LDS tile (row-major, stride LDT).
// ISA 16-bit A/B layout: lanes 0-15 hold rows 0-15 with K in {0..7,16..23},
// lanes 16-31 hold the same rows with K in {8..15,24..31}.
__device__ __forceinline__ v16bf load_frag_lds(const unsigned short* tileBase, int lane) {
    int row  = lane & 15;
    int half = (lane >> 4) << 3;   // 0 or 8
    const unsigned short* p = tileBase + row * LDT + half;
    union { uint4 q[2]; v16bf v; } f;
    f.q[0] = *(const uint4*)(p);
    f.q[1] = *(const uint4*)(p + 16);
    return f.v;
}

// 8 WMMAs for one wave's 64x32 patch from a staged LDS tile pair.
__device__ __forceinline__ void wmma_step(const unsigned short* As_,
                                          const unsigned short* Bs_,
                                          int wm, int wn, int lane,
                                          v8f (&acc)[4][2]) {
    v16bf af[4], bf[2];
    #pragma unroll
    for (int i = 0; i < 4; ++i)
        af[i] = load_frag_lds(As_ + (wm * 64 + i * 16) * LDT, lane);
    #pragma unroll
    for (int j = 0; j < 2; ++j)
        bf[j] = load_frag_lds(Bs_ + (wn * 32 + j * 16) * LDT, lane);
    #pragma unroll
    for (int i = 0; i < 4; ++i)
        #pragma unroll
        for (int j = 0; j < 2; ++j)
            acc[i][j] = __builtin_amdgcn_wmma_f32_16x16x32_bf16(
                false, af[i], false, bf[j], (short)0, acc[i][j], false, false);
}

// 16 consecutive fp32 held in registers (software pipeline stage).
struct Tile16 { float4 t[4]; };
__device__ __forceinline__ Tile16 load16(const float* p) {
    Tile16 r;
    const float4* p4 = (const float4*)p;
    r.t[0] = p4[0]; r.t[1] = p4[1]; r.t[2] = p4[2]; r.t[3] = p4[3];
    return r;
}
__device__ __forceinline__ void cvtst16(unsigned short* d, const Tile16& x) {
    #pragma unroll
    for (int q = 0; q < 4; ++q) {
        d[q * 4 + 0] = f2bf(x.t[q].x);
        d[q * 4 + 1] = f2bf(x.t[q].y);
        d[q * 4 + 2] = f2bf(x.t[q].z);
        d[q * 4 + 3] = f2bf(x.t[q].w);
    }
}

// ---------------------------------------------------------------------------
// Projection GEMM: Out(row r, col f) = X[r,:] . W[f,:] (+bias[f]) (*scale)
// X: fp32 (L*N, E) flattened from (l|s, n, e)  => global row g = l*16 + n
// W: fp32 (E_out=1024, E)
// MODE 0: Q -> bf16 [n][l][f], value = (acc+bias)*scale
// MODE 1: K -> bf16 [n][s][f]
// MODE 2: V -> bf16 transposed [n][f][s]
// Software-pipelined: tile k+1's global loads issue before tile k's WMMAs.
// ---------------------------------------------------------------------------
template <int MODE>
__global__ void __launch_bounds__(256)
proj_kernel(const float* __restrict__ X, const float* __restrict__ W,
            const float* __restrict__ bias, unsigned short* __restrict__ Out,
            float scale)
{
    __shared__ __align__(16) unsigned short As[BM * LDT];
    __shared__ __align__(16) unsigned short Bs[BN * LDT];

    const int tid  = threadIdx.x;
    const int lane = tid & 31;
    const int wave = tid >> 5;
    const int wm   = wave >> 2;   // 0..1 (64-row strip)
    const int wn   = wave & 3;    // 0..3 (32-col strip)

    const int row0 = blockIdx.x * BM;   // over 16384 rows
    const int col0 = blockIdx.y * BN;   // over 1024 out features

    // cooperative tile-load assignments: 128x32 fp32 -> bf16, 16 elems/thread
    const int arow = tid >> 1;          // 0..127
    const int aseg = (tid & 1) * 16;    // 0 / 16

    const float* aSrc = X + (size_t)(row0 + arow) * EDIM + aseg;
    const float* bSrc = W + (size_t)(col0 + arow) * EDIM + aseg;
    unsigned short* aDst = As + arow * LDT + aseg;
    unsigned short* bDst = Bs + arow * LDT + aseg;

    v8f acc[4][2] = {};

    Tile16 aR = load16(aSrc);
    Tile16 bR = load16(bSrc);

    for (int k0 = 0; k0 < EDIM; k0 += BK) {
        // convert + stage current tile (LDS free: end-barrier of prev iter)
        cvtst16(aDst, aR);
        cvtst16(bDst, bR);
        // issue next tile's loads before the WMMA block
        Tile16 aN = aR, bN = bR;
        if (k0 + BK < EDIM) {
            aN = load16(aSrc + k0 + BK);
            bN = load16(bSrc + k0 + BK);
            __builtin_prefetch(aSrc + k0 + 2 * BK, 0, 1);
            __builtin_prefetch(bSrc + k0 + 2 * BK, 0, 1);
        }
        __syncthreads();
        wmma_step(As, Bs, wm, wn, lane, acc);
        __syncthreads();
        aR = aN;
        bR = bN;
    }

    // Store. C layout: VGPR r, lane l -> M = r + (l>>4)*8, Ncol = l&15.
    // rBase is a multiple of 16, so global row g = rBase + M => n = M, l|s = rBase>>4.
    #pragma unroll
    for (int i = 0; i < 4; ++i) {
        const int rBase = row0 + wm * 64 + i * 16;
        const int seq   = rBase >> 4;              // l (Q) or s (K,V)
        #pragma unroll
        for (int j = 0; j < 2; ++j) {
            const int cBase = col0 + wn * 32 + j * 16;
            const int f     = cBase + (lane & 15);
            const float bv  = bias[f];
            #pragma unroll
            for (int r = 0; r < 8; ++r) {
                const int n = r + ((lane >> 4) << 3);   // batch index
                const float v = (acc[i][j][r] + bv) * scale;
                if (MODE == 2) {
                    Out[((size_t)n * EDIM + f) * SK + seq] = f2bf(v);   // Vt[n][f][s]
                } else {
                    Out[((size_t)n * LQ + seq) * EDIM + f] = f2bf(v);   // [n][l|s][f]
                }
            }
        }
    }
}

// ---------------------------------------------------------------------------
// Batched bf16 GEMM, K = 1024, per-batch A (1024 x 1024) and B-source
// (1024 rows x K contiguous). Tiles staged via DOUBLE-BUFFERED async
// global->LDS (ASYNCcnt): tile k+1 streams while tile k feeds the WMMAs.
// MODE 0: scores[n][l][s] = Q[n][l][:] . K[n][s][:]           (fp32 out)
// MODE 1: out[(l,n,f)]    = attn[n][l][:] . Vt[n][f][:]       (fp32 out)
// ---------------------------------------------------------------------------
template <int MODE>
__global__ void __launch_bounds__(256)
gemm_bf16_kernel(const unsigned short* __restrict__ Abuf,
                 const unsigned short* __restrict__ Bbuf,
                 float* __restrict__ Out)
{
    __shared__ __align__(16) unsigned short Sm[2][TILE_ELEMS];   // 2 x 20KB

    const int tid  = threadIdx.x;
    const int lane = tid & 31;
    const int wave = tid >> 5;
    const int wm   = wave >> 2;   // 0..1
    const int wn   = wave & 3;    // 0..3

    const int n    = blockIdx.z;
    const int row0 = blockIdx.x * BM;   // over L = 1024
    const int col0 = blockIdx.y * BN;   // over S or E = 1024

    const unsigned short* A = Abuf + (size_t)n * LQ * 1024;
    const unsigned short* B = Bbuf + (size_t)n * 1024 * 1024;

    // 128x32 bf16 tile = 8KB = 512 x 16B chunks; 2 chunks per thread per tile
    const int arow = tid >> 1;          // 0..127
    const int aseg = (tid & 1) * 16;    // 0 / 16  (elements)

    const unsigned short* aSrc = A + (size_t)(row0 + arow) * 1024 + aseg;
    const unsigned short* bSrc = B + (size_t)(col0 + arow) * 1024 + aseg;
    const unsigned aL0 = (unsigned)(size_t)(&Sm[0][arow * LDT + aseg]);
    const unsigned bL0 = (unsigned)(size_t)(&Sm[0][BM * LDT + arow * LDT + aseg]);

    v8f acc[4][2] = {};

    // Stage tile 0 into buffer 0.
    async_b128(aL0,      aSrc);
    async_b128(aL0 + 16, aSrc + 8);
    async_b128(bL0,      bSrc);
    async_b128(bL0 + 16, bSrc + 8);

    // 2x-unrolled double-buffered pipeline (32 K-steps total).
    for (int k0 = 0; k0 < 1024; k0 += 2 * BK) {
        // ---- step A: compute buf0 (tile k0), stage k0+32 into buf1 ----
        {
            const int kn = k0 + BK;
            if (kn < 1024) {
                async_b128(aL0 + TILE_BYTES,      aSrc + kn);
                async_b128(aL0 + TILE_BYTES + 16, aSrc + kn + 8);
                async_b128(bL0 + TILE_BYTES,      bSrc + kn);
                async_b128(bL0 + TILE_BYTES + 16, bSrc + kn + 8);
                wait_async4();
            } else {
                wait_async0();
            }
            __syncthreads();
            wmma_step(&Sm[0][0], &Sm[0][BM * LDT], wm, wn, lane, acc);
            __syncthreads();
        }
        // ---- step B: compute buf1 (tile k0+32), stage k0+64 into buf0 ----
        {
            const int kn = k0 + 2 * BK;
            if (kn < 1024) {
                async_b128(aL0,      aSrc + kn);
                async_b128(aL0 + 16, aSrc + kn + 8);
                async_b128(bL0,      bSrc + kn);
                async_b128(bL0 + 16, bSrc + kn + 8);
                wait_async4();
            } else {
                wait_async0();
            }
            __syncthreads();
            wmma_step(&Sm[1][0], &Sm[1][BM * LDT], wm, wn, lane, acc);
            __syncthreads();
        }
    }

    #pragma unroll
    for (int i = 0; i < 4; ++i) {
        const int rBase = row0 + wm * 64 + i * 16;   // l base
        #pragma unroll
        for (int j = 0; j < 2; ++j) {
            const int cBase = col0 + wn * 32 + j * 16;
            const int c     = cBase + (lane & 15);   // s (MODE 0) or f (MODE 1)
            #pragma unroll
            for (int r = 0; r < 8; ++r) {
                const int l = rBase + r + ((lane >> 4) << 3);
                if (MODE == 0) {
                    Out[((size_t)n * LQ + l) * SK + c] = acc[i][j][r];
                } else {
                    Out[((size_t)l * NBATCH + n) * EDIM + c] = acc[i][j][r];
                }
            }
        }
    }
}

// ---------------------------------------------------------------------------
// Row softmax over S=1024: one 256-thread block per (n,l) row; bf16 output.
// ---------------------------------------------------------------------------
__global__ void __launch_bounds__(256)
softmax_kernel(const float* __restrict__ scores, unsigned short* __restrict__ attn)
{
    __shared__ float red[8];
    const size_t row = blockIdx.x;
    const float* src = scores + row * SK;
    unsigned short* dst = attn + row * SK;
    const int tid  = threadIdx.x;
    const int lane = tid & 31;
    const int wave = tid >> 5;

    float v[4];
    float m = -3.402823466e38f;
    #pragma unroll
    for (int i = 0; i < 4; ++i) {
        v[i] = src[tid + i * 256];
        m = fmaxf(m, v[i]);
    }
    #pragma unroll
    for (int off = 16; off > 0; off >>= 1)
        m = fmaxf(m, __shfl_xor(m, off, 32));
    if (lane == 0) red[wave] = m;
    __syncthreads();
    m = red[0];
    #pragma unroll
    for (int w = 1; w < 8; ++w) m = fmaxf(m, red[w]);

    float s = 0.f;
    #pragma unroll
    for (int i = 0; i < 4; ++i) {
        v[i] = __expf(v[i] - m);
        s += v[i];
    }
    #pragma unroll
    for (int off = 16; off > 0; off >>= 1)
        s += __shfl_xor(s, off, 32);
    __syncthreads();                 // everyone done reading max values
    if (lane == 0) red[wave] = s;
    __syncthreads();
    s = 0.f;
    #pragma unroll
    for (int w = 0; w < 8; ++w) s += red[w];
    const float inv = 1.0f / s;

    #pragma unroll
    for (int i = 0; i < 4; ++i)
        dst[tid + i * 256] = f2bf(v[i] * inv);
}

// ---------------------------------------------------------------------------
extern "C" void kernel_launch(void* const* d_in, const int* in_sizes, int n_in,
                              void* d_out, int out_size, void* d_ws, size_t ws_size,
                              hipStream_t stream) {
    const float* query = (const float*)d_in[0];   // (L, N, E)
    const float* key   = (const float*)d_in[1];   // (S, N, E)
    const float* value = (const float*)d_in[2];   // (S, N, E)
    const float* wq    = (const float*)d_in[3];   // (E, E)
    const float* wk    = (const float*)d_in[4];
    const float* wv    = (const float*)d_in[5];
    const float* bias  = (const float*)d_in[6];   // (3E,)
    float* out = (float*)d_out;

    char* ws = (char*)d_ws;
    unsigned short* qb    = (unsigned short*)(ws);                        // 32 MiB  [n][l][e] bf16
    unsigned short* kb    = (unsigned short*)(ws + ((size_t)32  << 20));  // 32 MiB  [n][s][e] bf16
    unsigned short* vt    = (unsigned short*)(ws + ((size_t)64  << 20));  // 32 MiB  [n][f][s] bf16
    float*          sc    = (float*)         (ws + ((size_t)96  << 20));  // 64 MiB  [n][l][s] f32
    unsigned short* attnb = (unsigned short*)(ws + ((size_t)160 << 20));  // 32 MiB  [n][l][s] bf16

    const float scaling = 0.03125f;  // 1024^-0.5

    dim3 blk(256);
    dim3 gProj(16384 / BM, EDIM / BN);        // 128 x 8
    dim3 gGemm(LQ / BM, 1024 / BN, NBATCH);   // 8 x 8 x 16

    proj_kernel<0><<<gProj, blk, 0, stream>>>(query, wq, bias,            qb, scaling);
    proj_kernel<1><<<gProj, blk, 0, stream>>>(key,   wk, bias + EDIM,     kb, 1.0f);
    proj_kernel<2><<<gProj, blk, 0, stream>>>(value, wv, bias + 2 * EDIM, vt, 1.0f);

    gemm_bf16_kernel<0><<<gGemm, blk, 0, stream>>>(qb, kb, sc);
    softmax_kernel<<<dim3(NBATCH * LQ), blk, 0, stream>>>(sc, attnb);
    gemm_bf16_kernel<1><<<gGemm, blk, 0, stream>>>(attnb, vt, out);
}
